// EGNN_Model_31945966748215
// MI455X (gfx1250) — matrix operand order repaired and checked by reading, not compiled, not run on hardware
//
#include <hip/hip_runtime.h>
#include <hip/hip_bf16.h>

typedef _Float16 h16_t;
typedef __attribute__((ext_vector_type(16))) _Float16 v16h;
typedef __attribute__((ext_vector_type(8)))  _Float16 v8h;
typedef __attribute__((ext_vector_type(8)))  float    v8f;

#define BLK 256
#define MT  4            // M-tiles (16 rows each) per workgroup
#define C_NODES   20000
#define C_EDGES   320000
#define C_INDIM   11
#define C_HIDDEN  128
#define C_LAYERS  4
#define C_CLASSES 10
#define C_GRAPHS  64

// ---- fragment index helpers (per CDNA5 ISA 7.12.2 layouts) ----
// A-matrix 16x32 f16: lane group g (lane>>4), element e:
//   K = (e<8 ? 0 : 16) + 8*g + (e&7), M = lane&15
// => each half of the v16h (e=0..7 / e=8..15) covers 8 *contiguous* K values
//    starting at k0 = step*32 + hi*16 + g*8   (hi = e>=8)
__device__ __forceinline__ int kA_off(int g, int e) {
  return ((e < 8) ? 0 : 16) + 8 * g + (e & 7);
}
// B-matrix 32x16 f16: same K mapping as A, K = (e<8 ? 0 : 16) + 8*g + (e&7), N = lane&15

__device__ __forceinline__ v8f wmma_f16(v16h a, v16h b, v8f c) {
  return __builtin_amdgcn_wmma_f32_16x16x32_f16(false, a, false, b, (short)0, c,
                                                false, false);
}
__device__ __forceinline__ float silu_f(float v) { return v / (1.0f + __expf(-v)); }

// non-returning device-scope fp32 atomic add -> global_atomic_add_f32
__device__ __forceinline__ void atomic_add_f32(float* p, float v) {
  __hip_atomic_fetch_add(p, v, __ATOMIC_RELAXED, __HIP_MEMORY_SCOPE_AGENT);
}

__device__ __forceinline__ v16h ldfragA(const h16_t* A, int step, int lane) {
  return *(const v16h*)(A + step * 512 + lane * 16);
}
__device__ __forceinline__ v16h ldfragB(const h16_t* P, int step, int NT, int nt,
                                        int lane) {
  return *(const v16h*)(P + (size_t)(step * NT + nt) * 512 + lane * 16);
}
__device__ __forceinline__ v8h cvt8(float4 f0, float4 f1) {
  v8h v;
  v[0] = (h16_t)f0.x; v[1] = (h16_t)f0.y; v[2] = (h16_t)f0.z; v[3] = (h16_t)f0.w;
  v[4] = (h16_t)f1.x; v[5] = (h16_t)f1.y; v[6] = (h16_t)f1.z; v[7] = (h16_t)f1.w;
  return v;
}

// Write C fragment (+bias, SiLU) into an A-fragment-ordered LDS buffer (K = 128).
__device__ __forceinline__ void stage_hidden_silu(v8f c, const float* __restrict__ bias,
                                                  int wcol, h16_t* A2, int lane) {
  int g = lane >> 4, n15 = lane & 15;
  int n = wcol + n15;             // hidden column -> K of GEMM2 (wave-invariant bias)
  float bi = bias[n];
  int kl = n & 31;
  int g2 = (kl >> 3) & 1;
  int e  = ((kl & 16) ? 8 : 0) | (kl & 7);
  h16_t* base = A2 + (n >> 5) * 512 + e;
#pragma unroll
  for (int v = 0; v < 8; ++v) {
    int m = v + 8 * g;
    base[((g2 << 4) | m) << 4] = (h16_t)silu_f(c[v] + bi);
  }
}

// ---- weight pre-pack: fp32 [K,N] row-major -> f16 B-fragment order ----
__global__ void egnn_pack_w(const float* __restrict__ W, h16_t* __restrict__ P,
                            int K, int N, int S, int NT) {
  int total = S * NT * 512;
  for (int idx = blockIdx.x * blockDim.x + threadIdx.x; idx < total;
       idx += gridDim.x * blockDim.x) {
    int step = idx / (NT * 512);
    int r = idx - step * NT * 512;
    int nt = r >> 9;
    int q = r & 511;
    int lane = q >> 4, e = q & 15;
    int g = lane >> 4;
    int n = nt * 16 + (lane & 15);
    int k = step * 32 + ((e < 8) ? 0 : 16) + 8 * g + (e & 7);     // B-fragment K mapping
    P[idx] = (k < K && n < N) ? (h16_t)W[k * N + n] : (h16_t)0.0f;
  }
}

// fp32 h [N,11] -> f16 h [N,128] zero-padded
__global__ void egnn_cvt_h(const float* __restrict__ h, h16_t* __restrict__ h16) {
  int total = C_NODES * 128;
  for (int idx = blockIdx.x * blockDim.x + threadIdx.x; idx < total;
       idx += gridDim.x * blockDim.x) {
    int node = idx >> 7, k = idx & 127;
    h16[idx] = (k < C_INDIM) ? (h16_t)h[node * C_INDIM + k] : (h16_t)0.0f;
  }
}

// ---- fused edge MLP + scatter-add: 64 edges/block, 8 waves x 4 M-tiles ----
__global__ void egnn_edge_kernel(const float* __restrict__ x,
                                 const h16_t* __restrict__ hin,
                                 const int* __restrict__ src, const int* __restrict__ dst,
                                 const h16_t* __restrict__ W1, const float* __restrict__ b1,
                                 const h16_t* __restrict__ W2, const float* __restrict__ b2,
                                 float* __restrict__ aggr, int E, int d, int S1) {
  __shared__ __align__(32) h16_t A1[MT * 9 * 512];
  __shared__ __align__(32) h16_t A2[MT * 4 * 512];
  __shared__ int sdst[64], ssrc[64];
  __shared__ float sdist[64];
  const int tid = threadIdx.x;
  const int e0 = blockIdx.x << 6;
  if (tid < 64) {
    int e = e0 + tid;
    int s = 0, t = 0;
    float dd = 0.0f;
    if (e < E) {
      s = src[e]; t = dst[e];
#pragma unroll
      for (int j = 0; j < 3; ++j) {
        float df = x[t * 3 + j] - x[s * 3 + j];
        dd += df * df;
      }
    }
    ssrc[tid] = s; sdst[tid] = t; sdist[tid] = dd;
  }
  __syncthreads();
  // stage ef in A-frag order, 8-half (16B) groups; each group covers K=k0..k0+7
  const int d2 = 2 * d;
  for (int idx = tid; idx < MT * S1 * 64; idx += BLK) {
    int mt = idx / (S1 * 64);
    int rem = idx - mt * S1 * 64;
    int step = rem >> 6, gq = rem & 63;
    int lane = gq >> 1, hi = gq & 1;
    int g = lane >> 4;
    int m = (mt << 4) | (lane & 15);
    int k0 = (step << 5) + (hi << 4) + (g << 3);
    v8h vv = {};
    if (e0 + m < E) {
      if (k0 + 7 < d) {
        vv = *(const v8h*)(hin + sdst[m] * 128 + k0);                 // 16B aligned
      } else if (k0 >= d && k0 + 7 < d2 && (((k0 - d) & 7) == 0)) {
        vv = *(const v8h*)(hin + ssrc[m] * 128 + (k0 - d));
      } else if (k0 <= d2) {
#pragma unroll
        for (int j = 0; j < 8; ++j) {
          int k = k0 + j;
          h16_t v = (h16_t)0.0f;
          if (k < d)        v = hin[sdst[m] * 128 + k];
          else if (k < d2)  v = hin[ssrc[m] * 128 + (k - d)];
          else if (k == d2) v = (h16_t)sdist[m];
          vv[j] = v;
        }
      }
    }
    *(v8h*)(A1 + (((mt * S1 + step) << 9) | (lane << 4) | (hi << 3))) = vv;
  }
  __syncthreads();
  const int w = tid >> 5, lane = tid & 31;
  // GEMM1: one B fragment feeds 4 independent accumulators (no D->A/B hazards)
  v8f cc[MT] = {};
  for (int s = 0; s < S1; ++s) {
    v16h b = ldfragB(W1, s, 8, w, lane);
#pragma unroll
    for (int mt = 0; mt < MT; ++mt)
      cc[mt] = wmma_f16(ldfragA(A1, mt * S1 + s, lane), b, cc[mt]);
  }
#pragma unroll
  for (int mt = 0; mt < MT; ++mt)
    stage_hidden_silu(cc[mt], b1, w << 4, A2 + mt * 2048, lane);
  __syncthreads();
  v8f dd[MT] = {};
#pragma unroll
  for (int s = 0; s < 4; ++s) {
    v16h b = ldfragB(W2, s, 8, w, lane);
#pragma unroll
    for (int mt = 0; mt < MT; ++mt)
      dd[mt] = wmma_f16(ldfragA(A2 + mt * 2048, s, lane), b, dd[mt]);
  }
  const int g = lane >> 4, n15 = lane & 15;
  const int n = (w << 4) + n15;
  const float bi = b2[n];
#pragma unroll
  for (int mt = 0; mt < MT; ++mt) {
#pragma unroll
    for (int v = 0; v < 8; ++v) {
      int m = (mt << 4) + v + 8 * g;
      if (e0 + m < E)
        atomic_add_f32(&aggr[sdst[m] * 128 + n], silu_f(dd[mt][v] + bi));
    }
  }
}

// ---- fused node MLP: concat(h, aggr) -> new h (f16); 64 nodes/block ----
__global__ void egnn_node_kernel(const h16_t* __restrict__ hin,
                                 const float* __restrict__ aggr,
                                 const h16_t* __restrict__ W1, const float* __restrict__ b1,
                                 const h16_t* __restrict__ W2, const float* __restrict__ b2,
                                 h16_t* __restrict__ hout, int Nn, int d, int S1) {
  __shared__ __align__(32) h16_t A1[MT * 8 * 512];
  __shared__ __align__(32) h16_t A2[MT * 4 * 512];
  const int tid = threadIdx.x;
  const int n0 = blockIdx.x << 6;
  const int dk = d + 128;
  for (int idx = tid; idx < MT * S1 * 64; idx += BLK) {
    int mt = idx / (S1 * 64);
    int rem = idx - mt * S1 * 64;
    int step = rem >> 6, gq = rem & 63;
    int lane = gq >> 1, hi = gq & 1;
    int g = lane >> 4;
    int m = (mt << 4) | (lane & 15);
    int node = n0 + m;
    int k0 = (step << 5) + (hi << 4) + (g << 3);
    v8h vv = {};
    if (node < Nn) {
      if (k0 + 7 < d) {
        vv = *(const v8h*)(hin + node * 128 + k0);
      } else if (k0 >= d && k0 + 7 < dk && (((k0 - d) & 7) == 0)) {
        const float4* ap = (const float4*)(aggr + node * 128 + (k0 - d));
        vv = cvt8(ap[0], ap[1]);
      } else if (k0 < dk) {
#pragma unroll
        for (int j = 0; j < 8; ++j) {
          int k = k0 + j;
          h16_t v = (h16_t)0.0f;
          if (k < d)       v = hin[node * 128 + k];
          else if (k < dk) v = (h16_t)aggr[node * 128 + (k - d)];
          vv[j] = v;
        }
      }
    }
    *(v8h*)(A1 + (((mt * S1 + step) << 9) | (lane << 4) | (hi << 3))) = vv;
  }
  __syncthreads();
  const int w = tid >> 5, lane = tid & 31;
  v8f cc[MT] = {};
  for (int s = 0; s < S1; ++s) {
    v16h b = ldfragB(W1, s, 8, w, lane);
#pragma unroll
    for (int mt = 0; mt < MT; ++mt)
      cc[mt] = wmma_f16(ldfragA(A1, mt * S1 + s, lane), b, cc[mt]);
  }
#pragma unroll
  for (int mt = 0; mt < MT; ++mt)
    stage_hidden_silu(cc[mt], b1, w << 4, A2 + mt * 2048, lane);
  __syncthreads();
  v8f dd[MT] = {};
#pragma unroll
  for (int s = 0; s < 4; ++s) {
    v16h b = ldfragB(W2, s, 8, w, lane);
#pragma unroll
    for (int mt = 0; mt < MT; ++mt)
      dd[mt] = wmma_f16(ldfragA(A2 + mt * 2048, s, lane), b, dd[mt]);
  }
  const int g = lane >> 4, n15 = lane & 15;
  const int n = (w << 4) + n15;
  const float bi = b2[n];
#pragma unroll
  for (int mt = 0; mt < MT; ++mt) {
#pragma unroll
    for (int v = 0; v < 8; ++v) {
      int m = (mt << 4) + v + 8 * g;
      int node = n0 + m;
      if (node < Nn) hout[node * 128 + n] = (h16_t)(dd[mt][v] + bi);
    }
  }
}

// ---- final_mlp1 + global_add_pool (atomic into pooled); 64 nodes/block ----
__global__ void egnn_fm1_kernel(const h16_t* __restrict__ hin,
                                const int* __restrict__ batch,
                                const h16_t* __restrict__ W1, const float* __restrict__ b1,
                                const h16_t* __restrict__ W2, const float* __restrict__ b2,
                                float* __restrict__ pooled, int Nn) {
  __shared__ __align__(32) h16_t A1[MT * 4 * 512];
  __shared__ __align__(32) h16_t A2[MT * 4 * 512];
  __shared__ int sb[64];
  const int tid = threadIdx.x;
  const int n0 = blockIdx.x << 6;
  if (tid < 64) sb[tid] = (n0 + tid < Nn) ? batch[n0 + tid] : 0;
  for (int idx = tid; idx < MT * 4 * 64; idx += BLK) {   // K=128: always vector path
    int mt = idx >> 8;
    int rem = idx & 255;
    int step = rem >> 6, gq = rem & 63;
    int lane = gq >> 1, hi = gq & 1;
    int g = lane >> 4;
    int m = (mt << 4) | (lane & 15);
    int node = n0 + m;
    int k0 = (step << 5) + (hi << 4) + (g << 3);
    v8h vv = {};
    if (node < Nn) vv = *(const v8h*)(hin + node * 128 + k0);
    *(v8h*)(A1 + (((mt * 4 + step) << 9) | (lane << 4) | (hi << 3))) = vv;
  }
  __syncthreads();
  const int w = tid >> 5, lane = tid & 31;
  v8f cc[MT] = {};
#pragma unroll
  for (int s = 0; s < 4; ++s) {
    v16h b = ldfragB(W1, s, 8, w, lane);
#pragma unroll
    for (int mt = 0; mt < MT; ++mt)
      cc[mt] = wmma_f16(ldfragA(A1, mt * 4 + s, lane), b, cc[mt]);
  }
#pragma unroll
  for (int mt = 0; mt < MT; ++mt)
    stage_hidden_silu(cc[mt], b1, w << 4, A2 + mt * 2048, lane);
  __syncthreads();
  v8f dd[MT] = {};
#pragma unroll
  for (int s = 0; s < 4; ++s) {
    v16h b = ldfragB(W2, s, 8, w, lane);
#pragma unroll
    for (int mt = 0; mt < MT; ++mt)
      dd[mt] = wmma_f16(ldfragA(A2 + mt * 2048, s, lane), b, dd[mt]);
  }
  const int g = lane >> 4, n15 = lane & 15;
  const int n = (w << 4) + n15;
  const float bi = b2[n];
#pragma unroll
  for (int mt = 0; mt < MT; ++mt) {
#pragma unroll
    for (int v = 0; v < 8; ++v) {
      int m = (mt << 4) + v + 8 * g;
      if (n0 + m < Nn) atomic_add_f32(&pooled[sb[m] * 128 + n], dd[mt][v] + bi);
    }
  }
}

// ---- final_mlp2 on pooled [64,128] -> out [64,10]; 16 rows/block ----
__global__ void egnn_fm2_kernel(const float* __restrict__ pooled,
                                const h16_t* __restrict__ W1, const float* __restrict__ b1,
                                const h16_t* __restrict__ W2, const float* __restrict__ b2,
                                float* __restrict__ out) {
  __shared__ __align__(32) h16_t A1[4 * 512];
  __shared__ __align__(32) h16_t A2[4 * 512];
  const int tid = threadIdx.x;
  const int r0 = blockIdx.x << 4;
  for (int idx = tid; idx < 4 * 64; idx += BLK) {
    int step = idx >> 6, gq = idx & 63;
    int lane = gq >> 1, hi = gq & 1;
    int g = lane >> 4;
    int m = lane & 15;
    int k0 = (step << 5) + (hi << 4) + (g << 3);
    const float4* ap = (const float4*)(pooled + (r0 + m) * 128 + k0);
    *(v8h*)(A1 + ((step << 9) | (lane << 4) | (hi << 3))) = cvt8(ap[0], ap[1]);
  }
  __syncthreads();
  const int w = tid >> 5, lane = tid & 31;
  v8f c = {};
#pragma unroll
  for (int s = 0; s < 4; ++s)
    c = wmma_f16(ldfragA(A1, s, lane), ldfragB(W1, s, 8, w, lane), c);
  stage_hidden_silu(c, b1, w << 4, A2, lane);
  __syncthreads();
  if (w == 0) {            // single 16-col output tile (10 valid classes)
    v8f c2 = {};
#pragma unroll
    for (int s = 0; s < 4; ++s)
      c2 = wmma_f16(ldfragA(A2, s, lane), ldfragB(W2, s, 1, 0, lane), c2);
    const int g = lane >> 4, n15 = lane & 15;
#pragma unroll
    for (int v = 0; v < 8; ++v) {
      int m = v + 8 * g;
      if (n15 < C_CLASSES) out[(r0 + m) * C_CLASSES + n15] = c2[v] + b2[n15];
    }
  }
}

extern "C" void kernel_launch(void* const* d_in, const int* in_sizes, int n_in,
                              void* d_out, int out_size, void* d_ws, size_t ws_size,
                              hipStream_t stream) {
  (void)in_sizes; (void)n_in; (void)out_size; (void)ws_size;
  const float* x  = (const float*)d_in[0];
  const float* h0 = (const float*)d_in[1];
  const int* ei   = (const int*)d_in[2];
  const int* src  = ei;             // edge_index[0]
  const int* dst  = ei + C_EDGES;   // edge_index[1]
  const int* batch = (const int*)d_in[3];
  auto P32 = [&](int i) { return (const float*)d_in[i]; };

  // ---- workspace carve-out ----
  char* ws = (char*)d_ws;
  size_t cur = 0;
  auto alloc = [&](size_t bytes) -> char* {
    char* p = ws + cur;
    cur += (bytes + 255) & ~(size_t)255;
    return p;
  };
  h16_t* hA     = (h16_t*)alloc((size_t)C_NODES * 128 * 2);
  h16_t* hB     = (h16_t*)alloc((size_t)C_NODES * 128 * 2);
  float* aggr   = (float*)alloc((size_t)C_NODES * 128 * 4);
  float* pooled = (float*)alloc((size_t)C_GRAPHS * 128 * 4);

  h16_t *e1p[C_LAYERS], *e2p[C_LAYERS], *n1p[C_LAYERS], *n2p[C_LAYERS];
  int S_e1[C_LAYERS], S_n1[C_LAYERS], dl[C_LAYERS];
  for (int l = 0; l < C_LAYERS; ++l) {
    int d = (l == 0) ? C_INDIM : C_HIDDEN;
    dl[l] = d;
    S_e1[l] = (2 * d + 1 + 31) / 32;
    S_n1[l] = (d + 128 + 31) / 32;
    e1p[l] = (h16_t*)alloc((size_t)S_e1[l] * 8 * 512 * 2);
    e2p[l] = (h16_t*)alloc((size_t)4 * 8 * 512 * 2);
    n1p[l] = (h16_t*)alloc((size_t)S_n1[l] * 8 * 512 * 2);
    n2p[l] = (h16_t*)alloc((size_t)4 * 8 * 512 * 2);
  }
  h16_t* fm1ap = (h16_t*)alloc((size_t)4 * 8 * 512 * 2);
  h16_t* fm1bp = (h16_t*)alloc((size_t)4 * 8 * 512 * 2);
  h16_t* fm2ap = (h16_t*)alloc((size_t)4 * 8 * 512 * 2);
  h16_t* fm2bp = (h16_t*)alloc((size_t)4 * 1 * 512 * 2);

  // ---- pack weights into f16 B-fragment order (cheap, every call) ----
  auto pack = [&](const float* W, h16_t* P, int K, int N, int S, int NT) {
    int total = S * NT * 512;
    egnn_pack_w<<<(total + BLK - 1) / BLK, BLK, 0, stream>>>(W, P, K, N, S, NT);
  };
  for (int l = 0; l < C_LAYERS; ++l) {
    int base = 4 + l * 8;
    int d = dl[l];
    pack(P32(base + 0), e1p[l], 2 * d + 1, 128, S_e1[l], 8);
    pack(P32(base + 2), e2p[l], 128, 128, 4, 8);
    pack(P32(base + 4), n1p[l], d + 128, 128, S_n1[l], 8);
    pack(P32(base + 6), n2p[l], 128, 128, 4, 8);
  }
  pack(P32(36), fm1ap, 128, 128, 4, 8);
  pack(P32(38), fm1bp, 128, 128, 4, 8);
  pack(P32(40), fm2ap, 128, 128, 4, 8);
  pack(P32(42), fm2bp, 128, C_CLASSES, 4, 1);

  egnn_cvt_h<<<(C_NODES * 128 + BLK - 1) / BLK, BLK, 0, stream>>>(h0, hA);

  h16_t* hcur = hA;
  h16_t* hnext = hB;
  for (int l = 0; l < C_LAYERS; ++l) {
    int base = 4 + l * 8;
    hipMemsetAsync(aggr, 0, (size_t)C_NODES * 128 * 4, stream);
    egnn_edge_kernel<<<(C_EDGES + 63) / 64, BLK, 0, stream>>>(
        x, hcur, src, dst, e1p[l], P32(base + 1), e2p[l], P32(base + 3), aggr,
        C_EDGES, dl[l], S_e1[l]);
    egnn_node_kernel<<<(C_NODES + 63) / 64, BLK, 0, stream>>>(
        hcur, aggr, n1p[l], P32(base + 5), n2p[l], P32(base + 7), hnext,
        C_NODES, dl[l], S_n1[l]);
    h16_t* t = hcur; hcur = hnext; hnext = t;
  }
  hipMemsetAsync(pooled, 0, (size_t)C_GRAPHS * 128 * 4, stream);
  egnn_fm1_kernel<<<(C_NODES + 63) / 64, BLK, 0, stream>>>(
      hcur, batch, fm1ap, P32(37), fm1bp, P32(39), pooled, C_NODES);
  egnn_fm2_kernel<<<C_GRAPHS / 16, BLK, 0, stream>>>(
      pooled, fm2ap, P32(41), fm2bp, P32(43), (float*)d_out);
}
